// Net_44143673868650
// MI455X (gfx1250) — matrix operand order repaired and checked by reading, not compiled, not run on hardware
//
#include <hip/hip_runtime.h>

// CDNA5 / gfx1250 wave32 WMMA implementation of the SNN scan.
// GEMM weights (hi/lo f16 split of f32) are staged in LDS once and reused for
// all 100 timesteps; recurrent LIF state lives in VGPRs for the whole scan.
typedef __attribute__((ext_vector_type(16))) _Float16 v16h;
typedef __attribute__((ext_vector_type(8)))  float    v8f;

#define T_STEPS 100
#define BATCH   512
#define NINP    784
#define HID     1024
#define OUTD    10

#define BETA_   0.95f
#define DECAY_  0.05f   // DT / STDP_DECAY
#define APLUS_  0.01f
#define AMINUS_ 0.01f

#define KC1 25          // ceil(784/32) K-chunks for GEMM1 (padded to 800)
#define KD0 400         // dwords (f16 pairs) per (t,b) row of spkA0
#define KDV 392         // valid dwords (784/2)
#define KC2 32          // 1024/32 K-chunks for GEMM2

union ABfrag { uint4 q[2]; v16h h; };
union PackH  { _Float16 f[2]; unsigned int u; };

static __device__ __forceinline__ v8f vzero8() {
  v8f z;
#pragma unroll
  for (int i = 0; i < 8; ++i) z[i] = 0.0f;
  return z;
}

static __device__ __forceinline__ v8f wmma_f16(v16h a, v16h b, v8f c) {
  // D(f32 16x16) = A(f16 16x32) * B(f16 32x16) + C
  return __builtin_amdgcn_wmma_f32_16x16x32_f16(false, a, false, b, (short)0, c,
                                                false, false);
}

// ---------------------------------------------------------------------------
// Pack W1 (H x NIN, f32) into hi/lo f16 B-fragments.
// B-fragment layout (16-bit, 32x16): lanes 0-15 -> N=lane, K=0..7,16..23 (2/VGPR);
// lanes 16-31 -> N=lane-16, K=8..15,24..31. Storage: dword ((jt*KC1+kc)*32+lane)*8+v.
__global__ __launch_bounds__(256) void pack_w1(const float* __restrict__ W1,
                                               unsigned int* __restrict__ whi,
                                               unsigned int* __restrict__ wlo) {
  int idx  = blockIdx.x * 256 + threadIdx.x;   // < 64*25*256 = 409600
  int v    = idx & 7;
  int lane = (idx >> 3) & 31;
  int rest = idx >> 8;
  int kc   = rest % KC1;
  int jt   = rest / KC1;
  int h    = jt * 16 + (lane & 15);
  int k0   = kc * 32 + ((v >> 2) * 16) + ((lane >> 4) * 8) + 2 * (v & 3);
  float w0 = (k0     < NINP) ? W1[(size_t)h * NINP + k0]     : 0.0f;
  float w1 = (k0 + 1 < NINP) ? W1[(size_t)h * NINP + k0 + 1] : 0.0f;
  _Float16 h0 = (_Float16)w0, h1 = (_Float16)w1;
  _Float16 l0 = (_Float16)(w0 - (float)h0), l1 = (_Float16)(w1 - (float)h1);
  PackH ph; ph.f[0] = h0; ph.f[1] = h1;
  PackH pl; pl.f[0] = l0; pl.f[1] = l1;
  whi[idx] = ph.u;
  wlo[idx] = pl.u;
}

// Pack W2 (O x H) with N padded 10->16.
__global__ __launch_bounds__(256) void pack_w2(const float* __restrict__ W2,
                                               unsigned int* __restrict__ whi,
                                               unsigned int* __restrict__ wlo) {
  int idx  = blockIdx.x * 256 + threadIdx.x;   // < 32*256 = 8192
  int v    = idx & 7;
  int lane = (idx >> 3) & 31;
  int kc   = idx >> 8;
  int nn   = lane & 15;
  int k0   = kc * 32 + ((v >> 2) * 16) + ((lane >> 4) * 8) + 2 * (v & 3);   // always < 1024
  float w0 = (nn < OUTD) ? W2[(size_t)nn * HID + k0]     : 0.0f;
  float w1 = (nn < OUTD) ? W2[(size_t)nn * HID + k0 + 1] : 0.0f;
  _Float16 h0 = (_Float16)w0, h1 = (_Float16)w1;
  _Float16 l0 = (_Float16)(w0 - (float)h0), l1 = (_Float16)(w1 - (float)h1);
  PackH ph; ph.f[0] = h0; ph.f[1] = h1;
  PackH pl; pl.f[0] = l0; pl.f[1] = l1;
  whi[idx] = ph.u;
  wlo[idx] = pl.u;
}

// Zero the K-pad region (k = 784..799) of spkA0 for every (t,b).
__global__ __launch_bounds__(256) void zero_pads(unsigned int* __restrict__ spkA0) {
  int idx = blockIdx.x * 256 + threadIdx.x;    // < T*B*8 = 409600
  int row = idx >> 3;
  spkA0[(size_t)row * KD0 + KDV + (idx & 7)] = 0u;
}

// ---------------------------------------------------------------------------
// Layer 0: per-thread ownership of 2 input neurons across all T.
// thr=0 / subtract-reset => mem never resets: mem = BETA*mem + x; spk = mem>0.
__global__ __launch_bounds__(256) void layer0_scan(
    const float* __restrict__ x,
    float* __restrict__ spk0o, float* __restrict__ pre0o, float* __restrict__ post0o,
    unsigned int* __restrict__ spkA0) {
  int tid = blockIdx.x * 256 + threadIdx.x;    // exactly B*KDV = 200704
  int b   = tid / KDV;
  int kd  = tid % KDV;
  float m0 = 0.f, m1 = 0.f, p0 = 0.f, p1 = 0.f, q0 = 0.f, q1 = 0.f;
  for (int t = 0; t < T_STEPS; ++t) {
    size_t o = ((size_t)t * BATCH + b) * NINP + 2 * kd;
    float2 xv = *(const float2*)(x + o);
    m0 = BETA_ * m0 + xv.x;
    m1 = BETA_ * m1 + xv.y;
    float s0 = (m0 > 0.f) ? 1.f : 0.f;
    float s1 = (m1 > 0.f) ? 1.f : 0.f;
    p0 = (p0 - DECAY_ * p0) + APLUS_ * s0;
    p1 = (p1 - DECAY_ * p1) + APLUS_ * s1;
    q0 = (q0 - DECAY_ * q0) - AMINUS_ * s0;
    q1 = (q1 - DECAY_ * q1) - AMINUS_ * s1;
    *(float2*)(spk0o + o) = make_float2(s0, s1);
    *(float2*)(pre0o + o) = make_float2(p0, p1);
    *(float2*)(post0o + o) = make_float2(q0, q1);
    PackH ps; ps.f[0] = (_Float16)s0; ps.f[1] = (_Float16)s1;
    spkA0[((size_t)t * BATCH + b) * KD0 + kd] = ps.u;
  }
}

// ---------------------------------------------------------------------------
// GEMM1 + layer-1 LIF fused. 64 blocks x 8 waves. All 8 waves of a block share
// one hidden-tile pair jp; its W1 hi/lo fragments (100 KB) are staged in LDS
// once and reused for all 100 timesteps. Each wave owns a 32x32 (batch x
// hidden) output tile for all t; mem1/pre1/post1 stay in VGPRs. A-fragments
// (spikes) are prefetched one K-chunk ahead from L2.
__global__ __launch_bounds__(256) void gemm1_lif(
    const uint4* __restrict__ spkA0q,
    const uint4* __restrict__ w1hi, const uint4* __restrict__ w1lo,
    const float* __restrict__ thr1,
    float* __restrict__ spk1o, float* __restrict__ pre1o, float* __restrict__ post1o,
    _Float16* __restrict__ spkA1) {
  extern __shared__ char smem[];               // 2 * 51200 B (hi, lo)
  const int tid  = threadIdx.x;
  const int lane = tid & 31;
  const int wv   = tid >> 5;                   // wave in block 0..7
  const int jp   = blockIdx.x & 31;            // hidden tile pair 0..31
  const int ip   = ((blockIdx.x >> 5) << 3) + wv;  // batch tile pair 0..15
  const int n    = lane & 15;
  const int hi16 = lane >> 4;

  // One-time cooperative LDS fill: this block's W1 hi/lo fragment slice.
  {
    const size_t base4 = (size_t)(jp * 2) * KC1 * 64;   // uint4 index (=jt0*1600)
    uint4* dst = (uint4*)smem;
#pragma unroll 4
    for (int u = tid; u < 3200; u += 256) {
      dst[u]        = w1hi[base4 + u];
      dst[u + 3200] = w1lo[base4 + u];
    }
  }
  __syncthreads();
  const char* ldsH = smem;
  const char* ldsL = smem + 51200;

  v8f mem[4], pre[4], post[4];                 // index = ii*2 + jj
#pragma unroll
  for (int s = 0; s < 4; ++s) { mem[s] = vzero8(); pre[s] = vzero8(); post[s] = vzero8(); }
  const float thrj[2] = { thr1[(jp * 2 + 0) * 16 + n], thr1[(jp * 2 + 1) * 16 + n] };

  for (int t = 0; t < T_STEPS; ++t) {
    v8f c[4];
#pragma unroll
    for (int s = 0; s < 4; ++s) c[s] = vzero8();
    // A rows: lane holds row b = itile*16+n; lane-half selects K sub-range.
    const size_t ar0 = ((size_t)t * BATCH + ((ip * 2 + 0) * 16 + n)) * (KD0 / 4);
    const size_t ar1 = ((size_t)t * BATCH + ((ip * 2 + 1) * 16 + n)) * (KD0 / 4);
    ABfrag a0, a1;
    a0.q[0] = spkA0q[ar0 + hi16]; a0.q[1] = spkA0q[ar0 + hi16 + 2];
    a1.q[0] = spkA0q[ar1 + hi16]; a1.q[1] = spkA0q[ar1 + hi16 + 2];
    for (int kc = 0; kc < KC1; ++kc) {
      const int kn = (kc < KC1 - 1) ? kc + 1 : kc;   // prefetch next K-chunk
      ABfrag a0n, a1n;
      a0n.q[0] = spkA0q[ar0 + kn * 4 + hi16]; a0n.q[1] = spkA0q[ar0 + kn * 4 + hi16 + 2];
      a1n.q[0] = spkA0q[ar1 + kn * 4 + hi16]; a1n.q[1] = spkA0q[ar1 + kn * 4 + hi16 + 2];
#pragma unroll
      for (int jj = 0; jj < 2; ++jj) {
        const int fb = ((jj * KC1 + kc) * 32 + lane) * 32;   // frag byte offset
        ABfrag bh;
        bh.q[0] = *(const uint4*)(ldsH + fb);
        bh.q[1] = *(const uint4*)(ldsH + fb + 16);
        c[0 * 2 + jj] = wmma_f16(a0.h, bh.h, c[0 * 2 + jj]);
        c[1 * 2 + jj] = wmma_f16(a1.h, bh.h, c[1 * 2 + jj]);
        ABfrag bl;
        bl.q[0] = *(const uint4*)(ldsL + fb);
        bl.q[1] = *(const uint4*)(ldsL + fb + 16);
        c[0 * 2 + jj] = wmma_f16(a0.h, bl.h, c[0 * 2 + jj]);
        c[1 * 2 + jj] = wmma_f16(a1.h, bl.h, c[1 * 2 + jj]);
      }
      a0 = a0n; a1 = a1n;
    }
    // Fused layer-1 LIF + STDP traces, state in registers.
#pragma unroll
    for (int ii = 0; ii < 2; ++ii) {
#pragma unroll
      for (int jj = 0; jj < 2; ++jj) {
        const int s = ii * 2 + jj;
        const int hcol = (jp * 2 + jj) * 16 + n;
#pragma unroll
        for (int e = 0; e < 8; ++e) {
          float cur = c[s][e]; cur = cur > 0.f ? cur : 0.f;        // relu
          float mo = mem[s][e];
          float rs = (mo > thrj[jj]) ? 1.f : 0.f;                  // detached reset
          float mn = (BETA_ * mo + cur) * (1.f - rs);              // zero-reset
          float sp = ((mn - thrj[jj]) > 0.f) ? 1.f : 0.f;
          mem[s][e] = mn;
          float p = pre[s][e];  p = (p - DECAY_ * p) + APLUS_ * sp;  pre[s][e]  = p;
          float q = post[s][e]; q = (q - DECAY_ * q) - AMINUS_ * sp; post[s][e] = q;
          const int b = (ip * 2 + ii) * 16 + e + hi16 * 8;         // C-matrix M map
          const size_t o = ((size_t)t * BATCH + b) * HID + hcol;
          spk1o[o] = sp; pre1o[o] = p; post1o[o] = q;
          spkA1[o] = (_Float16)sp;                                 // same (t,b,h) layout
        }
      }
    }
  }
}

// ---------------------------------------------------------------------------
// GEMM2 + layer-2 LIF fused. One wave per 16-row batch tile, N padded 10->16.
// W2 hi/lo fragments (64 KB) staged in LDS once, reused for all timesteps.
__global__ __launch_bounds__(32) void gemm2_lif(
    const uint4* __restrict__ spkA1q,
    const uint4* __restrict__ w2hi, const uint4* __restrict__ w2lo,
    const float* __restrict__ thr2,
    float* __restrict__ spk2o, float* __restrict__ mem2o,
    float* __restrict__ pre2o, float* __restrict__ post2o) {
  extern __shared__ char smem[];        // 2 * 32768 B (hi, lo)
  const int lane = threadIdx.x & 31;
  const int i    = blockIdx.x;          // batch tile 0..31
  const int n    = lane & 15;
  const int hi16 = lane >> 4;

  {
    uint4* dst = (uint4*)smem;
#pragma unroll 4
    for (int u = lane; u < 2048; u += 32) {
      dst[u]        = w2hi[u];
      dst[u + 2048] = w2lo[u];
    }
  }
  __syncthreads();
  const char* ldsH = smem;
  const char* ldsL = smem + 32768;

  v8f mem = vzero8(), pre = vzero8(), post = vzero8();
  const float thr = (n < OUTD) ? thr2[n] : 1.0f;

  for (int t = 0; t < T_STEPS; ++t) {
    v8f ch = vzero8(), cl = vzero8();   // separate hi/lo chains
    const size_t arow4 = ((size_t)t * BATCH + (i * 16 + n)) * (HID / 2 / 4);
    ABfrag a;
    a.q[0] = spkA1q[arow4 + hi16]; a.q[1] = spkA1q[arow4 + hi16 + 2];
    for (int kc = 0; kc < KC2; ++kc) {
      const int kn = (kc < KC2 - 1) ? kc + 1 : kc;
      ABfrag an;
      an.q[0] = spkA1q[arow4 + kn * 4 + hi16];
      an.q[1] = spkA1q[arow4 + kn * 4 + hi16 + 2];
      const int fb = (kc * 32 + lane) * 32;
      ABfrag bh;
      bh.q[0] = *(const uint4*)(ldsH + fb);
      bh.q[1] = *(const uint4*)(ldsH + fb + 16);
      ch = wmma_f16(a.h, bh.h, ch);
      ABfrag bl;
      bl.q[0] = *(const uint4*)(ldsL + fb);
      bl.q[1] = *(const uint4*)(ldsL + fb + 16);
      cl = wmma_f16(a.h, bl.h, cl);
      a = an;
    }
#pragma unroll
    for (int e = 0; e < 8; ++e) {
      float cur = ch[e] + cl[e]; cur = cur > 0.f ? cur : 0.f;
      float mo = mem[e];
      float rs = (mo > thr) ? 1.f : 0.f;
      float mn = (BETA_ * mo + cur) * (1.f - rs);
      float sp = ((mn - thr) > 0.f) ? 1.f : 0.f;
      mem[e] = mn;
      float p = pre[e];  p = (p - DECAY_ * p) + APLUS_ * sp;  pre[e]  = p;
      float q = post[e]; q = (q - DECAY_ * q) - AMINUS_ * sp; post[e] = q;
      if (n < OUTD) {
        const int b = i * 16 + e + hi16 * 8;
        const size_t o = ((size_t)t * BATCH + b) * OUTD + n;
        spk2o[o] = sp; mem2o[o] = mn; pre2o[o] = p; post2o[o] = q;
      }
    }
  }
}

// ---------------------------------------------------------------------------
extern "C" void kernel_launch(void* const* d_in, const int* in_sizes, int n_in,
                              void* d_out, int out_size, void* d_ws, size_t ws_size,
                              hipStream_t stream) {
  const float* x    = (const float*)d_in[0];
  const float* W1   = (const float*)d_in[1];
  const float* W2   = (const float*)d_in[2];
  const float* thr1 = (const float*)d_in[3];
  const float* thr2 = (const float*)d_in[4];

  // Workspace layout (all 16B aligned). Total ~190.1 MB.
  char* ws = (char*)d_ws;
  const size_t W1PK_B  = (size_t)64 * KC1 * 32 * 8 * 4;       // 1,638,400 B each
  const size_t W2PK_B  = (size_t)KC2 * 32 * 8 * 4;            // 32,768 B each
  const size_t SPKA0_B = (size_t)T_STEPS * BATCH * KD0 * 4;   // 81,920,000 B
  unsigned int* w1hi  = (unsigned int*)(ws);
  unsigned int* w1lo  = (unsigned int*)(ws + W1PK_B);
  unsigned int* w2hi  = (unsigned int*)(ws + 2 * W1PK_B);
  unsigned int* w2lo  = (unsigned int*)(ws + 2 * W1PK_B + W2PK_B);
  unsigned int* spkA0 = (unsigned int*)(ws + 2 * W1PK_B + 2 * W2PK_B);
  _Float16*     spkA1 = (_Float16*)   (ws + 2 * W1PK_B + 2 * W2PK_B + SPKA0_B);

  // Output tuple (time-stacked, concatenated in return order).
  float* out = (float*)d_out;
  const size_t sz0 = (size_t)T_STEPS * BATCH * NINP;
  const size_t sz1 = (size_t)T_STEPS * BATCH * HID;
  const size_t sz2 = (size_t)T_STEPS * BATCH * OUTD;
  float* spk0o  = out;
  float* pre0o  = out + sz0;
  float* post0o = out + 2 * sz0;
  float* spk1o  = out + 3 * sz0;
  float* pre1o  = spk1o + sz1;
  float* post1o = spk1o + 2 * sz1;
  float* spk2o  = spk1o + 3 * sz1;
  float* mem2o  = spk2o + sz2;
  float* pre2o  = spk2o + 2 * sz2;
  float* post2o = spk2o + 3 * sz2;

  // Prep (every call: ws may be poisoned; all read regions are rewritten).
  pack_w1<<<1600, 256, 0, stream>>>(W1, w1hi, w1lo);
  pack_w2<<<32, 256, 0, stream>>>(W2, w2hi, w2lo);
  zero_pads<<<1600, 256, 0, stream>>>(spkA0);
  // Full-T scan, layer by layer (stream order gives the dependencies).
  layer0_scan<<<784, 256, 0, stream>>>(x, spk0o, pre0o, post0o, spkA0);
  gemm1_lif<<<64, 256, 2 * 51200, stream>>>((const uint4*)spkA0, (const uint4*)w1hi,
                                            (const uint4*)w1lo, thr1,
                                            spk1o, pre1o, post1o, spkA1);
  gemm2_lif<<<32, 32, 2 * 32768, stream>>>((const uint4*)spkA1, (const uint4*)w2hi,
                                           (const uint4*)w2lo, thr2,
                                           spk2o, mem2o, pre2o, post2o);
  (void)in_sizes; (void)n_in; (void)out_size; (void)ws_size;
}
